// def_att_4_13864154431694
// MI455X (gfx1250) — matrix operand order repaired and checked by reading, not compiled, not run on hardware
//
#include <hip/hip_runtime.h>
#include <hip/hip_bf16.h>

// ---------------------------------------------------------------------------
// Deformable attention forward for MI455X (gfx1250), wave32, WMMA-based.
// Heavy GEMMs: v_wmma_f32_16x16x32_bf16 (f32 accum); attn@V: v_wmma_f32_16x16x4_f32.
// Weights pre-packed in WMMA register order so every B-fragment is 2x b128.
// ---------------------------------------------------------------------------

typedef __attribute__((ext_vector_type(16))) __bf16 v16bf;
typedef __attribute__((ext_vector_type(8)))  float  v8f;
typedef __attribute__((ext_vector_type(2)))  float  v2f;

union BF16x16 { v16bf v; unsigned short u[16]; uint4 q4[2]; };

__device__ __forceinline__ unsigned short f2bf(float f) {
    unsigned u = __float_as_uint(f);
    u += 0x7FFFu + ((u >> 16) & 1u);            // round-to-nearest-even
    return (unsigned short)(u >> 16);
}
__device__ __forceinline__ float bf2f(unsigned short h) {
    return __uint_as_float(((unsigned)h) << 16);
}

// Workspace layout (bytes)
#define OFF_XN    ((size_t)0)            // bf16 [BT][64 pos][512 ch]
#define OFF_Q     ((size_t)67108864)     // bf16 [BT*G][64 pos][64 o]
#define OFF_VG    ((size_t)134217728)    // f32  [BT*G][4 key][2]
#define OFF_K     ((size_t)134479872)    // bf16 [BT*G][64 d][4 j]
#define OFF_V     ((size_t)138674176)    // f32  [BT*G][4 j][64 d]
#define OFF_BIAS  ((size_t)147062784)    // f32  [BT*G][64 i][4 j]
#define OFF_AO    ((size_t)155451392)    // bf16 [BT][64 pos][512 ch]
#define OFF_WQ    ((size_t)222560256)    // bf16 packed frags [g][nt4][ks2][lane][16]
#define OFF_W2T   ((size_t)222756864)    // bf16 packed frags [nt8][ks4][lane][16]
#define OFF_WOUT  ((size_t)222789632)    // bf16 packed frags [ks16][nt32][lane][16]

// ---------------------------------------------------------------------------
// K0: pack weights into bf16 WMMA B-fragment register order:
//     frag base + lane*16 + h, where n = nt*16+(lane&15), k = ks*32+(h&7)+(h>=8?16:0)+(lane>=16?8:0)
// ---------------------------------------------------------------------------
__global__ void k_prep(const float* __restrict__ wq, const float* __restrict__ w2,
                       const float* __restrict__ wout,
                       unsigned short* WQ, unsigned short* W2P, unsigned short* WOUTP) {
    int i = blockIdx.x * 256 + threadIdx.x;
    if (i < 32768) {                       // WQ: [g][nt(2b)][ks(1b)][lane(5b)][h(4b)]
        int h = i & 15, lane = (i >> 4) & 31, ks = (i >> 9) & 1, nt = (i >> 10) & 3, g = i >> 12;
        int n = nt * 16 + (lane & 15);
        int k = ks * 32 + (h & 7) + ((h >= 8) ? 16 : 0) + ((lane >= 16) ? 8 : 0);
        WQ[i] = f2bf(wq[g * 4096 + n * 64 + k]);       // wq[g][o=n][i=k]
    } else if (i < 49152) {                // W2P: [nt(3b)][ks(2b)][lane][h]
        int j = i - 32768;
        int h = j & 15, lane = (j >> 4) & 31, ks = (j >> 9) & 3, nt = j >> 11;
        int n = nt * 16 + (lane & 15);
        int k = ks * 32 + (h & 7) + ((h >= 8) ? 16 : 0) + ((lane >= 16) ? 8 : 0);
        W2P[j] = f2bf(w2[n * 128 + k]);                // cpb_w2[n][k]
    } else if (i < 311296) {               // WOUTP: [ks(4b)][nt(5b)][lane][h]
        int j = i - 49152;
        int h = j & 15, lane = (j >> 4) & 31, nt = (j >> 9) & 31, ks = j >> 14;
        int n = nt * 16 + (lane & 15);
        int k = ks * 32 + (h & 7) + ((h >= 8) ? 16 : 0) + ((lane >= 16) ? 8 : 0);
        WOUTP[j] = f2bf(wout[n * 512 + k]);            // w_out[o=n][c=k]
    }
}

// ---------------------------------------------------------------------------
// K1: LayerNorm over 512 ch + fold into 16 tiles of 8x8, bf16 out.
// Block = (batch n, image row): reads are full 128B image rows (coalesced),
// two passes over L2-hot data, padded-LDS staging, coalesced b128 stores.
// ---------------------------------------------------------------------------
__global__ void k_layernorm(const float* __restrict__ x, const float* __restrict__ lnw,
                            const float* __restrict__ lnb, unsigned short* __restrict__ xn) {
    int blk = blockIdx.x;                 // 2048 = 64 n * 32 rows
    int n = blk >> 5, row = blk & 31;
    int t = threadIdx.x;                  // 256 = 8 waves
    int lane = t & 31, w = t >> 5;
    __shared__ float sred[2][8][32];
    __shared__ unsigned short sxn[32][536];   // stride 536 halfs: 16B-aligned rows, low conflict
    const float* xrow = x + (size_t)n * 512 * 1024 + (size_t)row * 32;
    float s = 0.f, s2 = 0.f;
    for (int c = 0; c < 64; ++c) {
        int ch = w * 64 + c;
        float v = xrow[(size_t)ch * 1024 + lane];
        s += v; s2 += v * v;
    }
    sred[0][w][lane] = s; sred[1][w][lane] = s2;
    __syncthreads();
    float ts = 0.f, ts2 = 0.f;
#pragma unroll
    for (int ww = 0; ww < 8; ++ww) { ts += sred[0][ww][lane]; ts2 += sred[1][ww][lane]; }
    float mu = ts * (1.f / 512.f);
    float var = ts2 * (1.f / 512.f) - mu * mu;
    float rstd = rsqrtf(var + 1e-6f);
    for (int c = 0; c < 64; ++c) {
        int ch = w * 64 + c;
        float v = xrow[(size_t)ch * 1024 + lane];
        sxn[lane][ch] = f2bf((v - mu) * rstd * lnw[ch] + lnb[ch]);
    }
    __syncthreads();
    // cooperative coalesced store of the 32 pixels' 512-ch bf16 rows
    int th = row >> 3, pi = row & 7;
    int pix = t >> 3;                     // 0..31 (image col)
    int cb = (t & 7) * 64;                // 64-half chunk
    int tw = pix >> 3, pj = pix & 7;
    size_t dst = ((size_t)(th * 256 + tw * 64 + n) * 64 + (pi * 8 + pj)) * 512 + cb;
    const unsigned short* srcp = &sxn[pix][cb];
#pragma unroll
    for (int u = 0; u < 8; ++u)
        *(uint4*)(xn + dst + u * 8) = *(const uint4*)(srcp + u * 8);
}

// ---------------------------------------------------------------------------
// K2: q projection per (tile,group): q[64p][64o] = xn[64p][64i] @ wqT.
// A-frags: 2x b128 from xn rows; B-frags: 2x b128 from packed WQ.
// D staged via LDS, written back coalesced.
// ---------------------------------------------------------------------------
__global__ void k_qproj(const unsigned short* __restrict__ xn,
                        const unsigned short* __restrict__ WQ,
                        unsigned short* __restrict__ q) {
    int bg = blockIdx.x;
    int bt = bg >> 3, g = bg & 7;
    int lane = threadIdx.x & 31, mtile = threadIdx.x >> 5;   // 4 waves
    int m = lane & 15, hi8 = (lane >= 16) ? 8 : 0;
    __shared__ unsigned short sq[4][16][64];
    const unsigned short* arow = xn + ((size_t)bt * 64 + mtile * 16 + m) * 512 + g * 64;
    BF16x16 A[2];
#pragma unroll
    for (int ks = 0; ks < 2; ++ks) {
        const uint4* ap = (const uint4*)(arow + ks * 32 + hi8);
        A[ks].q4[0] = ap[0]; A[ks].q4[1] = ap[2];
    }
#pragma unroll
    for (int nt = 0; nt < 4; ++nt) {
        v8f acc = {};
#pragma unroll
        for (int ks = 0; ks < 2; ++ks) {
            const uint4* bp = (const uint4*)(WQ + (size_t)(((g * 4 + nt) * 2 + ks) * 32 + lane) * 16);
            BF16x16 Bf; Bf.q4[0] = bp[0]; Bf.q4[1] = bp[1];
            acc = __builtin_amdgcn_wmma_f32_16x16x32_bf16(false, A[ks].v, false, Bf.v,
                                                          (short)0, acc, false, false);
        }
#pragma unroll
        for (int r = 0; r < 8; ++r)
            sq[mtile][r + hi8][nt * 16 + m] = f2bf(acc[r]);
    }
    __syncthreads();
    unsigned short* qd = q + (size_t)bg * 4096;
    int p = threadIdx.x >> 1;             // 0..63
    int cb = (threadIdx.x & 1) * 32;
    const unsigned short* srcp = &sq[p >> 4][p & 15][cb];
#pragma unroll
    for (int u = 0; u < 4; ++u)
        *(uint4*)(qd + p * 64 + cb + u * 8) = *(const uint4*)(srcp + u * 8);
}

// ---------------------------------------------------------------------------
// K3: offset net: depthwise 6x6 s4 p1 -> GELU -> 1x1 -> tanh*4 -> sample grid
// ---------------------------------------------------------------------------
__global__ void k_offsets(const unsigned short* __restrict__ q,
                          const float* __restrict__ dww, const float* __restrict__ dwb,
                          const float* __restrict__ pw, float* __restrict__ vgrid) {
    int bg = blockIdx.x;
    int c = threadIdx.x;                  // channel 0..63
    __shared__ float st[64][4];
    float img[64];
    const unsigned short* qb = q + (size_t)bg * 4096;
#pragma unroll
    for (int p = 0; p < 64; ++p) img[p] = bf2f(qb[p * 64 + c]);
#pragma unroll
    for (int oy = 0; oy < 2; ++oy)
#pragma unroll
        for (int ox = 0; ox < 2; ++ox) {
            float t = 0.f;
#pragma unroll
            for (int ky = 0; ky < 6; ++ky) {
                int iy = oy * 4 - 1 + ky;
                if (iy < 0 || iy >= 8) continue;
#pragma unroll
                for (int kx = 0; kx < 6; ++kx) {
                    int ix = ox * 4 - 1 + kx;
                    if (ix < 0 || ix >= 8) continue;
                    t += img[iy * 8 + ix] * dww[c * 36 + ky * 6 + kx];
                }
            }
            t += dwb[c];
            t = 0.5f * t * (1.0f + erff(t * 0.70710678118654752f));   // exact GELU
            st[c][oy * 2 + ox] = t;
        }
    __syncthreads();
    if (c < 4) {
        int p = c, px = p & 1, py = p >> 1;
        float s0 = 0.f, s1 = 0.f;
        for (int i = 0; i < 64; ++i) { float t = st[i][p]; s0 += pw[i] * t; s1 += pw[64 + i] * t; }
        float vx = (float)px + tanhf(s0) * 4.0f;
        float vy = (float)py + tanhf(s1) * 4.0f;
        vgrid[bg * 8 + p * 2 + 0] = 2.f * vx - 1.f;
        vgrid[bg * 8 + p * 2 + 1] = 2.f * vy - 1.f;
    }
}

// ---------------------------------------------------------------------------
// K4: bilinear grid-sample of 4 keys + k/v projections (tiny GEMV, VALU).
// ---------------------------------------------------------------------------
__global__ void k_kv(const unsigned short* __restrict__ xn, const float* __restrict__ vgrid,
                     const float* __restrict__ wk, const float* __restrict__ wv,
                     unsigned short* __restrict__ Kb, float* __restrict__ Vb) {
    int bg = blockIdx.x;
    int bt = bg >> 3, g = bg & 7;
    __shared__ float skv[4][64];
    int t = threadIdx.x;                  // 256
    {
        int j = t >> 6, i = t & 63;
        float gxn = vgrid[bg * 8 + j * 2 + 0];
        float gyn = vgrid[bg * 8 + j * 2 + 1];
        float ix = ((gxn + 1.f) * 8.f - 1.f) * 0.5f;
        float iy = ((gyn + 1.f) * 8.f - 1.f) * 0.5f;
        float ix0f = floorf(ix), iy0f = floorf(iy);
        float wx1 = ix - ix0f, wy1 = iy - iy0f, wx0 = 1.f - wx1, wy0 = 1.f - wy1;
        int x0 = (int)ix0f, y0 = (int)iy0f;
        const unsigned short* xb = xn + (size_t)bt * 64 * 512 + g * 64 + i;
        auto Gs = [&](int xa, int ya) -> float {
            if (xa < 0 || xa > 7 || ya < 0 || ya > 7) return 0.f;
            return bf2f(xb[(ya * 8 + xa) * 512]);
        };
        skv[j][i] = Gs(x0, y0) * wx0 * wy0 + Gs(x0 + 1, y0) * wx1 * wy0 +
                    Gs(x0, y0 + 1) * wx0 * wy1 + Gs(x0 + 1, y0 + 1) * wx1 * wy1;
    }
    __syncthreads();
    int o = t >> 2, jj = t & 3;
    const float* wkg = wk + (size_t)(g * 64 + o) * 64;
    const float* wvg = wv + (size_t)(g * 64 + o) * 64;
    float sk = 0.f, sv = 0.f;
#pragma unroll 8
    for (int ii = 0; ii < 64; ++ii) {
        float kvv = skv[jj][ii];
        sk += wkg[ii] * kvv;
        sv += wvg[ii] * kvv;
    }
    Kb[(size_t)bg * 256 + o * 4 + jj] = f2bf(sk);   // [d][j]
    Vb[(size_t)bg * 256 + jj * 64 + o] = sv;        // [j][d]
}

// ---------------------------------------------------------------------------
// K5: CPB MLP bias (the FLOP king). layer1 VALU -> LDS bf16; layer2 WMMA bf16
// (A-frags 2x ds b128, B-frags 2x global b128 from packed W2P); layer3 via
// LDS f32 atomics.
// ---------------------------------------------------------------------------
__global__ void k_cpb(const float* __restrict__ vgrid, const float* __restrict__ w1,
                      const float* __restrict__ b1, const unsigned short* __restrict__ W2P,
                      const float* __restrict__ b2, const float* __restrict__ w3,
                      const float* __restrict__ b3, float* __restrict__ bias) {
    int blk = blockIdx.x;
    int bg = blk >> 1, half = blk & 1;
    __shared__ unsigned short sh1[128][128];
    __shared__ float sacc[128];
    int t = threadIdx.x;                  // 256 = 8 waves
    {
        int r = t >> 1, cb = (t & 1) * 64;
        int ig = half * 32 + (r >> 2), j = r & 3;
        float qx = 2.f * (float)(ig & 7) / 7.f - 1.f;
        float qy = 2.f * (float)(ig >> 3) / 7.f - 1.f;
        float kx = vgrid[bg * 8 + j * 2 + 0];
        float ky = vgrid[bg * 8 + j * 2 + 1];
        float p0 = qx - kx, p1 = qy - ky;
        float s0 = copysignf(log1pf(fabsf(p0)), p0);
        float s1 = copysignf(log1pf(fabsf(p1)), p1);
        for (int cc = 0; cc < 64; ++cc) {
            int col = cb + cc;
            float h = b1[col] + s0 * w1[col * 2] + s1 * w1[col * 2 + 1];
            sh1[r][col] = f2bf(fmaxf(h, 0.f));
        }
        if (t < 128) sacc[t] = 0.f;
    }
    __syncthreads();
    int lane = t & 31, mtile = t >> 5;
    int m = lane & 15, hi8 = (lane >= 16) ? 8 : 0;
    BF16x16 A4[4];
#pragma unroll
    for (int ks = 0; ks < 4; ++ks) {
        const uint4* ap = (const uint4*)&sh1[mtile * 16 + m][ks * 32 + hi8];
        A4[ks].q4[0] = ap[0]; A4[ks].q4[1] = ap[2];
    }
#pragma unroll
    for (int nt = 0; nt < 8; ++nt) {
        v8f acc = {};
#pragma unroll
        for (int ks = 0; ks < 4; ++ks) {
            const uint4* bp = (const uint4*)(W2P + (size_t)(((nt * 4 + ks) * 32 + lane) * 16));
            BF16x16 Bf; Bf.q4[0] = bp[0]; Bf.q4[1] = bp[1];
            acc = __builtin_amdgcn_wmma_f32_16x16x32_bf16(false, A4[ks].v, false, Bf.v,
                                                          (short)0, acc, false, false);
        }
        int n = nt * 16 + m;
        float w3n = w3[n], b2n = b2[n];
#pragma unroll
        for (int r = 0; r < 8; ++r) {
            int row = mtile * 16 + r + hi8;
            float h2 = fmaxf(acc[r] + b2n, 0.f);
            atomicAdd(&sacc[row], h2 * w3n);     // ds_add_f32
        }
    }
    __syncthreads();
    if (t < 128) {
        int ig = half * 32 + (t >> 2), j = t & 3;
        bias[(size_t)bg * 256 + ig * 4 + j] = sacc[t] + b3[0];
    }
}

// ---------------------------------------------------------------------------
// K6: sim (WMMA bf16, N padded 4->16) + bias + softmax + attn@V (WMMA f32 K=4)
// ---------------------------------------------------------------------------
__global__ void k_attn(const unsigned short* __restrict__ qb, const unsigned short* __restrict__ Kb,
                       const float* __restrict__ Vb, const float* __restrict__ bias,
                       unsigned short* __restrict__ ao) {
    int bg = blockIdx.x;
    int bt = bg >> 3, g = bg & 7;
    __shared__ float sat[64][4];
    __shared__ unsigned short sao[64][64];
    int t = threadIdx.x;                  // 128 = 4 waves
    int lane = t & 31, mtile = t >> 5;
    int m = lane & 15, hi8 = (lane >= 16) ? 8 : 0, hi16 = (lane >= 16) ? 16 : 0;
    const unsigned short* qd = qb + (size_t)bg * 4096;
    const unsigned short* kd = Kb + (size_t)bg * 256;
    v8f acc = {};
#pragma unroll
    for (int ks = 0; ks < 2; ++ks) {
        BF16x16 Af, Bf;
        const uint4* ap = (const uint4*)(qd + (mtile * 16 + m) * 64 + ks * 32 + hi8);
        Af.q4[0] = ap[0]; Af.q4[1] = ap[2];
#pragma unroll
        for (int h = 0; h < 16; ++h) {
            int kb = ks * 32 + (h & 7) + ((h >= 8) ? 16 : 0) + hi8;
            Bf.u[h] = (m < 4) ? kd[kb * 4 + m] : (unsigned short)0;
        }
        acc = __builtin_amdgcn_wmma_f32_16x16x32_bf16(false, Af.v, false, Bf.v,
                                                      (short)0, acc, false, false);
    }
    if (m < 4) {
#pragma unroll
        for (int r = 0; r < 8; ++r) {
            int i = mtile * 16 + r + hi8;
            sat[i][m] = acc[r] * 0.125f + bias[(size_t)bg * 256 + i * 4 + m];  // SCALE=1/8
        }
    }
    __syncthreads();
    if (t < 64) {
        float a0 = sat[t][0], a1 = sat[t][1], a2 = sat[t][2], a3 = sat[t][3];
        float mx = fmaxf(fmaxf(a0, a1), fmaxf(a2, a3));
        float e0 = expf(a0 - mx), e1 = expf(a1 - mx), e2 = expf(a2 - mx), e3 = expf(a3 - mx);
        float rs = 1.f / (e0 + e1 + e2 + e3);
        sat[t][0] = e0 * rs; sat[t][1] = e1 * rs; sat[t][2] = e2 * rs; sat[t][3] = e3 * rs;
    }
    __syncthreads();
    // attn(64x4) @ v^T(4x64) with v_wmma_f32_16x16x4_f32
    int k0 = (lane >= 16) ? 2 : 0;
    v2f Aa;
    Aa[0] = sat[mtile * 16 + m][k0];
    Aa[1] = sat[mtile * 16 + m][k0 + 1];
    const float* vd = Vb + (size_t)bg * 256;
#pragma unroll
    for (int nt = 0; nt < 4; ++nt) {
        int d = nt * 16 + m;
        v2f Bb;
        Bb[0] = vd[k0 * 64 + d];
        Bb[1] = vd[(k0 + 1) * 64 + d];
        v8f c = {};
        c = __builtin_amdgcn_wmma_f32_16x16x4_f32(false, Aa, false, Bb, (short)0, c, false, false);
#pragma unroll
        for (int r = 0; r < 8; ++r)
            sao[mtile * 16 + r + hi8][d] = f2bf(c[r]);
    }
    __syncthreads();
    unsigned short* aob = ao + (size_t)bt * 64 * 512 + g * 64;
    int p = t >> 1, cb = (t & 1) * 32;
    const unsigned short* srcp = &sao[p][cb];
#pragma unroll
    for (int u = 0; u < 4; ++u)
        *(uint4*)(aob + (size_t)p * 512 + cb + u * 8) = *(const uint4*)(srcp + u * 8);
}

// ---------------------------------------------------------------------------
// K7: output projection ao(64x512) @ w_outT(512x512) + bias, un-fold tiles.
// No LDS: A-frags 2x b128 from ao rows; B-frags 2x b128 from packed WOUTP.
// ---------------------------------------------------------------------------
__global__ void __launch_bounds__(512)
k_outproj(const unsigned short* __restrict__ ao, const unsigned short* __restrict__ WOUT,
          const float* __restrict__ bout, float* __restrict__ out) {
    int bt = blockIdx.x;
    int t = threadIdx.x;                  // 512 = 16 waves
    int lane = t & 31, w = t >> 5;
    int mtile = w & 3, nq = w >> 2;       // 4 m-tiles x 4 n-quads (8 n-tiles each)
    int m = lane & 15, hi8 = (lane >= 16) ? 8 : 0;
    v8f acc[8];
#pragma unroll
    for (int nt = 0; nt < 8; ++nt) acc[nt] = (v8f){};
    const unsigned short* arow = ao + ((size_t)bt * 64 + mtile * 16 + m) * 512;
    for (int ks = 0; ks < 16; ++ks) {
        BF16x16 Af;
        const uint4* ap = (const uint4*)(arow + ks * 32 + hi8);
        Af.q4[0] = ap[0]; Af.q4[1] = ap[2];
        if (ks < 15)
            __builtin_prefetch(WOUT + (size_t)((ks + 1) * 32) * 512, 0, 1);
#pragma unroll
        for (int nt = 0; nt < 8; ++nt) {
            int ntile = nq * 8 + nt;
            const uint4* bp = (const uint4*)(WOUT + (size_t)(((ks * 32 + ntile) * 32 + lane) * 16));
            BF16x16 Bf; Bf.q4[0] = bp[0]; Bf.q4[1] = bp[1];
            acc[nt] = __builtin_amdgcn_wmma_f32_16x16x32_bf16(false, Af.v, false, Bf.v,
                                                              (short)0, acc[nt], false, false);
        }
    }
    int n = bt & 63, tw = (bt >> 6) & 3, th = bt >> 8;
#pragma unroll
    for (int nt = 0; nt < 8; ++nt) {
        int ch = (nq * 8 + nt) * 16 + m;
        float bb = bout[ch];
#pragma unroll
        for (int r = 0; r < 8; ++r) {
            int pos = mtile * 16 + r + hi8;
            int pi = pos >> 3, pj = pos & 7;
            out[(((size_t)n * 512 + ch) * 32 + th * 8 + pi) * 32 + tw * 8 + pj] = acc[nt][r] + bb;
        }
    }
}

// ---------------------------------------------------------------------------
extern "C" void kernel_launch(void* const* d_in, const int* in_sizes, int n_in,
                              void* d_out, int out_size, void* d_ws, size_t ws_size,
                              hipStream_t stream) {
    (void)in_sizes; (void)n_in; (void)out_size; (void)ws_size;
    const float* x      = (const float*)d_in[0];
    const float* ln_w   = (const float*)d_in[1];
    const float* ln_b   = (const float*)d_in[2];
    const float* wq     = (const float*)d_in[3];
    const float* wk     = (const float*)d_in[4];
    const float* wv     = (const float*)d_in[5];
    const float* w_out  = (const float*)d_in[6];
    const float* b_out  = (const float*)d_in[7];
    const float* dw_w   = (const float*)d_in[8];
    const float* dw_b   = (const float*)d_in[9];
    const float* pw_w   = (const float*)d_in[10];
    const float* cw1    = (const float*)d_in[11];
    const float* cb1    = (const float*)d_in[12];
    const float* cw2    = (const float*)d_in[13];
    const float* cb2    = (const float*)d_in[14];
    const float* cw3    = (const float*)d_in[15];
    const float* cb3    = (const float*)d_in[16];
    float* out = (float*)d_out;

    char* ws = (char*)d_ws;
    unsigned short* XN    = (unsigned short*)(ws + OFF_XN);
    unsigned short* Q     = (unsigned short*)(ws + OFF_Q);
    float*          VG    = (float*)(ws + OFF_VG);
    unsigned short* KB    = (unsigned short*)(ws + OFF_K);
    float*          VB    = (float*)(ws + OFF_V);
    float*          BIAS  = (float*)(ws + OFF_BIAS);
    unsigned short* AO    = (unsigned short*)(ws + OFF_AO);
    unsigned short* WQb   = (unsigned short*)(ws + OFF_WQ);
    unsigned short* W2P   = (unsigned short*)(ws + OFF_W2T);
    unsigned short* WOUTb = (unsigned short*)(ws + OFF_WOUT);

    k_prep<<<1216, 256, 0, stream>>>(wq, cw2, w_out, WQb, W2P, WOUTb);
    k_layernorm<<<2048, 256, 0, stream>>>(x, ln_w, ln_b, XN);
    k_qproj<<<8192, 128, 0, stream>>>(XN, WQb, Q);
    k_offsets<<<8192, 64, 0, stream>>>(Q, dw_w, dw_b, pw_w, VG);
    k_kv<<<8192, 256, 0, stream>>>(XN, VG, wk, wv, KB, VB);
    k_cpb<<<16384, 256, 0, stream>>>(VG, cw1, cb1, W2P, cb2, cw3, cb3, BIAS);
    k_attn<<<8192, 128, 0, stream>>>(Q, KB, VB, BIAS, AO);
    k_outproj<<<1024, 512, 0, stream>>>(AO, WOUTb, b_out, out);
}
